// ChunkRnn_59536836657848
// MI455X (gfx1250) — compile-verified, matrix operand order checked by reading, not run on hardware
//
#include <hip/hip_runtime.h>
#include <hip/hip_bf16.h>

// Problem constants (from reference): B=4, N=8192, D=1024, H=16, C=128, HD=64
#define BB   4
#define NN   8192
#define DDIM 1024
#define HH   16
#define CC   128
#define HDD  64
#define NCH  64            // N / C
#define MTOT (BB * NN)     // 32768 rows
#define KSTEP 64           // K-step per LDS tile in the big GEMM

typedef __attribute__((ext_vector_type(16))) __bf16 bf16x16;
typedef __attribute__((ext_vector_type(8)))  __bf16 bf16x8;
typedef __attribute__((ext_vector_type(8)))  float  f32x8;
typedef __attribute__((ext_vector_type(4)))  int    i32x4;

// CDNA5 async global->LDS path (ASYNCcnt), with sync fallback if the
// toolchain doesn't expose the builtins.
#if defined(__has_builtin)
#  if __has_builtin(__builtin_amdgcn_global_load_async_to_lds_b128) && \
      __has_builtin(__builtin_amdgcn_s_wait_asynccnt)
#    define HAVE_ASYNC_LDS 1
#  endif
#endif
#ifndef HAVE_ASYNC_LDS
#  define HAVE_ASYNC_LDS 0
#endif

// The builtin's params are AS1 / AS3 pointers to int32x4 (per hipcc diagnostic).
typedef __attribute__((address_space(1))) i32x4 as1_i32x4;
typedef __attribute__((address_space(3))) i32x4 as3_i32x4;

#if HAVE_ASYNC_LDS
#  define ASYNC_WAIT(n) __builtin_amdgcn_s_wait_asynccnt(n)
#else
#  define ASYNC_WAIT(n) ((void)0)
#endif

__device__ __forceinline__ f32x8 zero8() {
    f32x8 z = {0.f, 0.f, 0.f, 0.f, 0.f, 0.f, 0.f, 0.f};
    return z;
}

// Load a 16x32 bf16 A-fragment (or 32x16 B-fragment) for wave32 WMMA.
// ISA layout (16-bit A 16x32): lane l holds row m=l&15; k-half kh=l>>4;
// elems 0..7 -> k = kh*8 + 0..7 ; elems 8..15 -> k = 16 + kh*8 + 0..7.
__device__ __forceinline__ bf16x16 load_frag_row(const __bf16* base, int row, int ld,
                                                 int kbase, int lane) {
    const int kh = (lane >> 4) & 1;
    const __bf16* p = base + (size_t)row * ld + kbase + kh * 8;
    union U { bf16x16 v; bf16x8 h[2]; } u;
    u.h[0] = *(const bf16x8*)(p);
    u.h[1] = *(const bf16x8*)(p + 16);
    return u.v;
}

__device__ __forceinline__ f32x8 wmma_bf16(bf16x16 a, bf16x16 b, f32x8 c) {
    return __builtin_amdgcn_wmma_f32_16x16x32_bf16(false, a, false, b, (short)0, c,
                                                   false, false);
}

// Cooperative stage of a 64-row x KSTEP-col W tile into LDS.
// 256 threads x 2 ops x 16B = 8KB. Async when available (2 ASYNCcnt per wave).
__device__ __forceinline__ void stage_B_tile(const __bf16* __restrict__ W, int K,
                                             int n0, int kb,
                                             __bf16 (*sB)[KSTEP], int tid) {
#pragma unroll
    for (int o = 0; o < 2; ++o) {
        const int flat = tid + o * 256;
        const int row = flat >> 3;          // 8 threads per 64-elem row
        const int ko = (flat & 7) * 8;      // 16B chunks
        const __bf16* g = W + (size_t)(n0 + row) * K + kb + ko;
        __bf16* l = &sB[row][ko];
#if HAVE_ASYNC_LDS
        __builtin_amdgcn_global_load_async_to_lds_b128((as1_i32x4*)g, (as3_i32x4*)l, 0, 0);
#else
        *(bf16x8*)l = *(const bf16x8*)g;
#endif
    }
}

// ---------------------------------------------------------------------------
// fp32 -> bf16 conversion (grid-stride)
// ---------------------------------------------------------------------------
__global__ void cvt_f32_bf16_kernel(const float* __restrict__ in,
                                    __bf16* __restrict__ out, size_t n) {
    size_t i = (size_t)blockIdx.x * blockDim.x + threadIdx.x;
    size_t stride = (size_t)gridDim.x * blockDim.x;
    for (; i < n; i += stride) out[i] = (__bf16)in[i];
}

// ---------------------------------------------------------------------------
// GEMM: Y[M,N] = act(X[M,K] @ W[N,K]^T), bf16 in, f32 accumulate.
// W tile (64 x KSTEP) double-buffered in LDS via async-to-LDS; A fragments in
// registers. Wave tile 32x64 (2 A-frags x 4 B-frags). Block = 8 waves =
// 256x64 output. grid = (M/256, N/64). act: 0 = none, 1 = silu.
// ---------------------------------------------------------------------------
__global__ __launch_bounds__(256) void gemm_bf16_nt_kernel(
    const __bf16* __restrict__ X, const __bf16* __restrict__ W,
    float* __restrict__ Yf, __bf16* __restrict__ Yb,
    int M, int N, int K, int act) {
    __shared__ __bf16 sB[2][64][KSTEP];     // 16 KB double buffer

    const int tid = threadIdx.x;
    const int lane = tid & 31;
    const int wv = tid >> 5;
    const int m0 = blockIdx.x * 256 + wv * 32;
    const int n0 = blockIdx.y * 64;
    const int nl = lane & 15;
    const int arow0 = m0 + nl;
    const int arow1 = m0 + 16 + nl;

    f32x8 acc0[4], acc1[4];
#pragma unroll
    for (int t = 0; t < 4; ++t) { acc0[t] = zero8(); acc1[t] = zero8(); }

    const int nk = K / KSTEP;
    stage_B_tile(W, K, n0, 0, sB[0], tid);
    int p = 0;

    for (int it = 0; it < nk; ++it) {
        const int kb = it * KSTEP;
        if (it + 1 < nk) {
            stage_B_tile(W, K, n0, kb + KSTEP, sB[1 - p], tid);
            ASYNC_WAIT(2);      // current tile retired (in-order), next in flight
        } else {
            ASYNC_WAIT(0);
        }
        __syncthreads();

        // A fragments for both 32-k substeps, both 16-row groups
        bf16x16 a00 = load_frag_row(X, arow0, K, kb, lane);
        bf16x16 a01 = load_frag_row(X, arow0, K, kb + 32, lane);
        bf16x16 a10 = load_frag_row(X, arow1, K, kb, lane);
        bf16x16 a11 = load_frag_row(X, arow1, K, kb + 32, lane);

#pragma unroll
        for (int t = 0; t < 4; ++t) {
            bf16x16 b0 = load_frag_row(&sB[p][0][0], t * 16 + nl, KSTEP, 0, lane);
            acc0[t] = wmma_bf16(a00, b0, acc0[t]);
            acc1[t] = wmma_bf16(a10, b0, acc1[t]);
            bf16x16 b1 = load_frag_row(&sB[p][0][0], t * 16 + nl, KSTEP, 32, lane);
            acc0[t] = wmma_bf16(a01, b1, acc0[t]);
            acc1[t] = wmma_bf16(a11, b1, acc1[t]);
        }
        __syncthreads();        // before buffer p is refilled next iteration
        p ^= 1;
    }

    const int rh = (lane >> 4) * 8;   // C/D layout: m = r + 8*(lane>>4)
#pragma unroll
    for (int t = 0; t < 4; ++t) {
#pragma unroll
        for (int r = 0; r < 8; ++r) {
            float v0 = acc0[t][r];
            float v1 = acc1[t][r];
            if (act) {
                v0 = v0 / (1.0f + __expf(-v0));   // silu
                v1 = v1 / (1.0f + __expf(-v1));
            }
            const size_t i0 = (size_t)(m0 + rh + r) * N + (n0 + t * 16 + nl);
            const size_t i1 = (size_t)(m0 + 16 + rh + r) * N + (n0 + t * 16 + nl);
            if (Yf) { Yf[i0] = v0; Yf[i1] = v1; }
            if (Yb) { Yb[i0] = (__bf16)v0; Yb[i1] = (__bf16)v1; }
        }
    }
}

// ---------------------------------------------------------------------------
// Gate projection: log_f[row,h] = logsigmoid(x[row,:] . Wf[h,:] + delta[h])
// ---------------------------------------------------------------------------
__global__ __launch_bounds__(128) void gate_kernel(
    const float* __restrict__ X, const float* __restrict__ Wf,
    const float* __restrict__ delta, float* __restrict__ LG) {
    __shared__ float sx[DDIM];
    __shared__ float red[HH][4];
    const int row = blockIdx.x;
    for (int i = threadIdx.x; i < DDIM; i += 128)
        sx[i] = X[(size_t)row * DDIM + i];
    __syncthreads();
    const int lane = threadIdx.x & 31, wv = threadIdx.x >> 5;
    for (int h = 0; h < HH; ++h) {
        float s = 0.f;
        for (int i = threadIdx.x; i < DDIM; i += 128)
            s += sx[i] * Wf[(size_t)h * DDIM + i];
        for (int off = 16; off > 0; off >>= 1) s += __shfl_down(s, off);
        if (lane == 0) red[h][wv] = s;
    }
    __syncthreads();
    if (threadIdx.x < HH) {
        const int h = threadIdx.x;
        float z = red[h][0] + red[h][1] + red[h][2] + red[h][3] + delta[h];
        float ls = fminf(z, 0.0f) - log1pf(__expf(-fabsf(z)));  // log_sigmoid
        LG[(size_t)row * HH + h] = ls;
    }
}

// ---------------------------------------------------------------------------
// In-chunk inclusive cumsum of log gates. grid = B*H*NC, block = C (128).
// LG layout [B*N, H]; LF layout [B, H, NC, C].
// ---------------------------------------------------------------------------
__global__ __launch_bounds__(128) void cumsum_kernel(
    const float* __restrict__ LG, float* __restrict__ LF) {
    __shared__ float buf[2][CC];
    const int g = blockIdx.x;
    const int b = g >> 10;
    const int h = (g >> 6) & 15;
    const int ci = g & 63;
    const int t = threadIdx.x;
    buf[0][t] = LG[((size_t)b * NN + (size_t)ci * CC + t) * HH + h];
    __syncthreads();
    int src = 0;
    for (int off = 1; off < CC; off <<= 1) {
        float x = buf[src][t];
        float add = (t >= off) ? buf[src][t - off] : 0.0f;
        buf[1 - src][t] = x + add;
        src = 1 - src;
        __syncthreads();
    }
    LF[(size_t)g * CC + t] = buf[src][t];
}

// ---------------------------------------------------------------------------
// Chunked gated-attention scan. One 256-thread block (8 waves) per (b,h).
// ---------------------------------------------------------------------------
__global__ __launch_bounds__(256) void scan_kernel(
    const __bf16* __restrict__ Q, const __bf16* __restrict__ Km,
    const __bf16* __restrict__ V, const float* __restrict__ LF,
    float* __restrict__ O) {
    __shared__ float  sS[HDD][HDD];       // state, f32           16 KB
    __shared__ __bf16 sSbT[HDD][HDD];     // S^T bf16 (B-operand)  8 KB
    __shared__ __bf16 sA[CC][CC];         // masked/decayed A     32 KB
    __shared__ __bf16 sVT[HDD][CC];       // v^T  (B-operand)     16 KB
    __shared__ __bf16 sKT[HDD][CC];       // (k*decay)^T          16 KB
    __shared__ float  sLb[CC];

    const int bh = blockIdx.x;
    const int b = bh >> 4;
    const int h = bh & 15;
    const int lane = threadIdx.x & 31;
    const int wv = threadIdx.x >> 5;
    const int tid = threadIdx.x;
    const int nl = lane & 15;
    const int rh = (lane >> 4) * 8;

    for (int i = tid; i < HDD * HDD; i += 256) {
        ((float*)sS)[i] = 0.0f;
        ((__bf16*)sSbT)[i] = (__bf16)0.0f;
    }
    __syncthreads();

    const size_t rowbase = (size_t)b * NN;

    for (int ci = 0; ci < NCH; ++ci) {
        // --- 1. per-chunk cumulative log-gates ---
        if (tid < CC)
            sLb[tid] = LF[(((size_t)(b * HH + h) * NCH + ci) * CC) + tid];
        __syncthreads();
        const float tot = sLb[CC - 1];
        const float etot = __expf(tot);

        // --- 2. stage v^T and decayed-k^T in LDS ---
        for (int f = tid; f < CC * HDD; f += 256) {
            const int i = f >> 6;
            const int d = f & 63;
            const size_t g = (rowbase + (size_t)ci * CC + i) * DDIM + h * HDD + d;
            sVT[d][i] = V[g];
            sKT[d][i] = (__bf16)((float)Km[g] * __expf(tot - sLb[i]));
        }
        __syncthreads();

        // --- 3. A = q k^T (128x128, K=64); wave wv owns rows [wv*16, +16) ---
        {
            const int m0 = wv * 16;
            f32x8 accA[8];
#pragma unroll
            for (int t = 0; t < 8; ++t) accA[t] = zero8();
            const __bf16* qrow = Q + (rowbase + (size_t)ci * CC) * DDIM + h * HDD;
            const __bf16* krow = Km + (rowbase + (size_t)ci * CC) * DDIM + h * HDD;
#pragma unroll
            for (int kb = 0; kb < HDD; kb += 32) {
                bf16x16 a = load_frag_row(qrow, m0 + nl, DDIM, kb, lane);
#pragma unroll
                for (int t = 0; t < 8; ++t) {
                    bf16x16 bb = load_frag_row(krow, t * 16 + nl, DDIM, kb, lane);
                    accA[t] = wmma_bf16(a, bb, accA[t]);
                }
            }
#pragma unroll
            for (int t = 0; t < 8; ++t) {
                const int j = t * 16 + nl;
#pragma unroll
                for (int r = 0; r < 8; ++r) {
                    const int i = m0 + rh + r;
                    float v = (j <= i) ? accA[t][r] * __expf(sLb[i] - sLb[j]) : 0.0f;
                    sA[i][j] = (__bf16)v;
                }
            }
        }
        __syncthreads();

        // --- 4. o = (q * exp(l)) @ S + A @ v   (128x64) ---
        {
            const int m0 = wv * 16;
            f32x8 accO[4];
#pragma unroll
            for (int t = 0; t < 4; ++t) accO[t] = zero8();

            const float qs = __expf(sLb[m0 + nl]);
            const __bf16* qrow = Q + (rowbase + (size_t)ci * CC) * DDIM + h * HDD;
#pragma unroll
            for (int kb = 0; kb < HDD; kb += 32) {
                bf16x16 a0 = load_frag_row(qrow, m0 + nl, DDIM, kb, lane);
                bf16x16 a;
#pragma unroll
                for (int e = 0; e < 16; ++e) a[e] = (__bf16)((float)a0[e] * qs);
#pragma unroll
                for (int t = 0; t < 4; ++t) {
                    bf16x16 bb = load_frag_row(&sSbT[0][0], t * 16 + nl, HDD, kb, lane);
                    accO[t] = wmma_bf16(a, bb, accO[t]);
                }
            }
#pragma unroll
            for (int kb = 0; kb < CC; kb += 32) {
                bf16x16 a = load_frag_row(&sA[0][0], m0 + nl, CC, kb, lane);
#pragma unroll
                for (int t = 0; t < 4; ++t) {
                    bf16x16 bb = load_frag_row(&sVT[0][0], t * 16 + nl, CC, kb, lane);
                    accO[t] = wmma_bf16(a, bb, accO[t]);
                }
            }
#pragma unroll
            for (int t = 0; t < 4; ++t) {
#pragma unroll
                for (int r = 0; r < 8; ++r) {
                    const int i = m0 + rh + r;
                    const int d = t * 16 + nl;
                    O[(rowbase + (size_t)ci * CC + i) * DDIM + h * HDD + d] = accO[t][r];
                }
            }
        }

        // --- 5. state update: S = exp(tot)*S + kdecay^T @ v  (64x64, K=128) ---
        {
            f32x8 accU[2];
#pragma unroll
            for (int u = 0; u < 2; ++u) {
                accU[u] = zero8();
                const int t = wv * 2 + u;
                const int m0 = (t >> 2) * 16;
                const int n0 = (t & 3) * 16;
#pragma unroll
                for (int kb = 0; kb < CC; kb += 32) {
                    bf16x16 a = load_frag_row(&sKT[0][0], m0 + nl, CC, kb, lane);
                    bf16x16 bb = load_frag_row(&sVT[0][0], n0 + nl, CC, kb, lane);
                    accU[u] = wmma_bf16(a, bb, accU[u]);
                }
            }
            __syncthreads();
#pragma unroll
            for (int u = 0; u < 2; ++u) {
                const int t = wv * 2 + u;
                const int m0 = (t >> 2) * 16;
                const int n0 = (t & 3) * 16;
#pragma unroll
                for (int r = 0; r < 8; ++r) {
                    const int dd = m0 + rh + r;
                    const int ee = n0 + nl;
                    const float s = etot * sS[dd][ee] + accU[u][r];
                    sS[dd][ee] = s;
                    sSbT[ee][dd] = (__bf16)s;
                }
            }
        }
        __syncthreads();
    }
}

// ---------------------------------------------------------------------------
// RMSNorm row-wise, f32 in -> bf16 out.
// ---------------------------------------------------------------------------
__global__ __launch_bounds__(256) void rmsnorm_kernel(
    const float* __restrict__ O, const float* __restrict__ gamma,
    __bf16* __restrict__ Y) {
    __shared__ float red[8];
    const int row = blockIdx.x;
    const float* p = O + (size_t)row * DDIM;
    float ss = 0.f;
    for (int i = threadIdx.x; i < DDIM; i += 256) {
        float v = p[i];
        ss += v * v;
    }
    for (int off = 16; off > 0; off >>= 1) ss += __shfl_down(ss, off);
    if ((threadIdx.x & 31) == 0) red[threadIdx.x >> 5] = ss;
    __syncthreads();
    if (threadIdx.x == 0) {
        float t = 0.f;
        for (int i = 0; i < 8; ++i) t += red[i];
        red[0] = rsqrtf(t / (float)DDIM + 1e-6f);
    }
    __syncthreads();
    const float sc = red[0];
    for (int i = threadIdx.x; i < DDIM; i += 256)
        Y[(size_t)row * DDIM + i] = (__bf16)(p[i] * sc * gamma[i]);
}

// ---------------------------------------------------------------------------
// Host launcher
// ---------------------------------------------------------------------------
extern "C" void kernel_launch(void* const* d_in, const int* in_sizes, int n_in,
                              void* d_out, int out_size, void* d_ws, size_t ws_size,
                              hipStream_t stream) {
    const float* x  = (const float*)d_in[0];
    const float* Wq = (const float*)d_in[1];
    const float* Wk = (const float*)d_in[2];
    const float* Wv = (const float*)d_in[3];
    const float* Wo = (const float*)d_in[4];
    const float* Wf = (const float*)d_in[5];
    const float* delta = (const float*)d_in[6];
    const float* gamma = (const float*)d_in[7];
    float* out = (float*)d_out;

    char* ws = (char*)d_ws;
    size_t off = 0;
    auto alloc = [&](size_t bytes) {
        char* p = ws + off;
        off += (bytes + 255) & ~(size_t)255;
        return p;
    };
    const size_t nX = (size_t)MTOT * DDIM;
    const size_t nW = (size_t)DDIM * DDIM;
    __bf16* xb = (__bf16*)alloc(nX * 2);     // x bf16; later reused for o_norm
    __bf16* wq = (__bf16*)alloc(nW * 2);
    __bf16* wk = (__bf16*)alloc(nW * 2);
    __bf16* wv = (__bf16*)alloc(nW * 2);
    __bf16* wo = (__bf16*)alloc(nW * 2);
    __bf16* qb = (__bf16*)alloc(nX * 2);
    __bf16* kb = (__bf16*)alloc(nX * 2);
    __bf16* vb = (__bf16*)alloc(nX * 2);
    float* lg = (float*)alloc((size_t)MTOT * HH * 4);
    float* lf = (float*)alloc((size_t)BB * HH * NCH * CC * 4);
    float* of = (float*)alloc(nX * 4);
    (void)ws_size; (void)n_in; (void)in_sizes; (void)out_size;

    // 1. fp32 -> bf16 conversions
    cvt_f32_bf16_kernel<<<2048, 256, 0, stream>>>(x, xb, nX);
    cvt_f32_bf16_kernel<<<1024, 256, 0, stream>>>(Wq, wq, nW);
    cvt_f32_bf16_kernel<<<1024, 256, 0, stream>>>(Wk, wk, nW);
    cvt_f32_bf16_kernel<<<1024, 256, 0, stream>>>(Wv, wv, nW);
    cvt_f32_bf16_kernel<<<1024, 256, 0, stream>>>(Wo, wo, nW);

    // 2. q/k/v projections (WMMA); silu on q,k
    dim3 gg(MTOT / 256, DDIM / 64);
    gemm_bf16_nt_kernel<<<gg, 256, 0, stream>>>(xb, wq, nullptr, qb, MTOT, DDIM, DDIM, 1);
    gemm_bf16_nt_kernel<<<gg, 256, 0, stream>>>(xb, wk, nullptr, kb, MTOT, DDIM, DDIM, 1);
    gemm_bf16_nt_kernel<<<gg, 256, 0, stream>>>(xb, wv, nullptr, vb, MTOT, DDIM, DDIM, 0);

    // 3. gates + in-chunk cumsum
    gate_kernel<<<MTOT, 128, 0, stream>>>(x, Wf, delta, lg);
    cumsum_kernel<<<BB * HH * NCH, 128, 0, stream>>>(lg, lf);

    // 4. chunked scan (WMMA), one block per (b,h)
    scan_kernel<<<BB * HH, 256, 0, stream>>>(qb, kb, vb, lf, of);

    // 5. RMSNorm (bf16 out, reusing xb) + output projection (WMMA, f32 out)
    rmsnorm_kernel<<<MTOT, 256, 0, stream>>>(of, gamma, xb);
    gemm_bf16_nt_kernel<<<gg, 256, 0, stream>>>(xb, wo, out, nullptr, MTOT, DDIM, DDIM, 0);
}
